// BioSNN_53953379173274
// MI455X (gfx1250) — matrix operand order repaired and checked by reading, not compiled, not run on hardware
//
#include <hip/hip_runtime.h>
#include <hip/hip_bf16.h>
#include <stdint.h>

#define T_STEPS 128
#define BATCH   128
#define NIN     2048
#define NH      4096
#define NO      512
#define NREC    (NH + NO)      // 4608
#define KT1     (NIN / 32)     // 64 K-tiles, layer 1
#define KT2     (NREC / 32)    // 144 K-tiles, layer 2
#define CHUNK   48             // K-tiles staged per LDS phase in layer 2

typedef __attribute__((ext_vector_type(16))) __bf16 v16bf;
typedef __attribute__((ext_vector_type(8)))  float  v8f;
typedef __attribute__((address_space(3))) uint16_t lds_u16;

union Frag16 {
  v16bf v;
  uint4 u[2];
};

// async global->LDS 16B transfer (per lane), ISA GLOBAL_LOAD_ASYNC_TO_LDS_B128
__device__ __forceinline__ void async_copy16(uint32_t lds_off, const void* g) {
  asm volatile("global_load_async_to_lds_b128 %0, %1, off"
               :: "v"(lds_off), "v"(g) : "memory");
}
__device__ __forceinline__ void wait_async0() {
  asm volatile("s_wait_asynccnt 0x0" ::: "memory");
}

// round-to-nearest-even f32 -> bf16 (bit pattern)
__device__ __forceinline__ uint16_t f2bf(float f) {
  union { float f; uint32_t u; } c; c.f = f;
  uint32_t u = c.u;
  uint32_t r = u + 0x7FFFu + ((u >> 16) & 1u);
  return (uint16_t)(r >> 16);
}

// ---------------------------------------------------------------- convert x
__global__ void k_f32_to_bf16(const float* __restrict__ src,
                              uint16_t* __restrict__ dst, int n) {
  int i = blockIdx.x * blockDim.x + threadIdx.x;
  int stride = gridDim.x * blockDim.x;
  for (; i < n; i += stride) dst[i] = f2bf(src[i]);
}

// ------------------------------------------- pack W (row-major [N, K]) into
// WMMA B-fragment tiles: dst[((nt*ktiles + kt)*32 + lane)*16 + h]
//   column N = nt*16 + lane%16
//   K        = kt*32 + (h&7) + (lane>=16 ? 8 : 0) + (h>=8 ? 16 : 0)   (ISA 7.12.2 B layout, wave32)
__global__ void k_pack_b(const float* __restrict__ W, uint16_t* __restrict__ dst,
                         int ktiles, int total, int ldw) {
  int p = blockIdx.x * blockDim.x + threadIdx.x;
  if (p >= total) return;
  int h    = p & 15;
  int L    = (p >> 4) & 31;
  int rest = p >> 9;
  int kt   = rest % ktiles;
  int nt   = rest / ktiles;
  int n = nt * 16 + (L & 15);
  int k = kt * 32 + (h & 7) + ((L & 16) ? 8 : 0) + ((h & 8) ? 16 : 0);
  dst[p] = f2bf(W[(size_t)n * ldw + k]);
}

// ---------------------------------------------------------------- init state
__global__ void k_init(float* __restrict__ syn2, float* __restrict__ mem2,
                       uint32_t* __restrict__ spk2buf2, float* __restrict__ memtr) {
  int i = blockIdx.x * blockDim.x + threadIdx.x;
  if (i < BATCH * NO) { syn2[i] = 0.f; mem2[i] = 0.f; spk2buf2[i] = 0u; }
  if (i < T_STEPS) memtr[i] = 0.f;
}

// -------------------------------------- phase 1: cur1_all = X @ W1^T + b1
// X is [T*B, NIN] bf16 row-major; one wave computes a 16(M) x 64(N) block.
// Register double-buffered: next-kt fragments are in flight while current
// fragments feed the WMMA pipe.
__global__ void __launch_bounds__(256)
k_gemm1(const uint16_t* __restrict__ xbf, const uint16_t* __restrict__ w1p,
        const float* __restrict__ b1, float* __restrict__ cur1) {
  int wave  = blockIdx.x * 8 + (threadIdx.x >> 5); // 0..65535
  int lane  = threadIdx.x & 31;
  int mtile = wave >> 6;   // 0..1023  (16 rows each)
  int nquad = wave & 63;   // 0..63    (64 cols each)
  int m0 = mtile * 16;
  int lm = lane & 15;
  int hh = (lane >> 4) & 1;

  const uint16_t* arow = xbf + (size_t)(m0 + lm) * NIN;

  v8f acc[4];
#pragma unroll
  for (int q = 0; q < 4; ++q)
#pragma unroll
    for (int r = 0; r < 8; ++r) acc[q][r] = 0.0f;

  Frag16 aC, aN, bC[4], bN[4];
  {
    const uint4* pa = (const uint4*)(arow + hh * 8);
    aC.u[0] = pa[0]; aC.u[1] = pa[2];
#pragma unroll
    for (int q = 0; q < 4; ++q) {
      const uint4* pb = (const uint4*)(w1p + (((size_t)(nquad * 4 + q) * KT1) * 32 + lane) * 16);
      bC[q].u[0] = pb[0]; bC[q].u[1] = pb[1];
    }
  }
  for (int kt = 0; kt < KT1; ++kt) {
    int ktn = (kt + 1 < KT1) ? kt + 1 : kt;
    const uint4* pa = (const uint4*)(arow + ktn * 32 + hh * 8);
    aN.u[0] = pa[0]; aN.u[1] = pa[2];
#pragma unroll
    for (int q = 0; q < 4; ++q) {
      const uint4* pb = (const uint4*)(w1p + (((size_t)(nquad * 4 + q) * KT1 + ktn) * 32 + lane) * 16);
      bN[q].u[0] = pb[0]; bN[q].u[1] = pb[1];
    }
#pragma unroll
    for (int q = 0; q < 4; ++q)
      acc[q] = __builtin_amdgcn_wmma_f32_16x16x32_bf16(
          false, aC.v, false, bC[q].v, (short)0, acc[q], false, false);
    aC = aN;
#pragma unroll
    for (int q = 0; q < 4; ++q) bC[q] = bN[q];
  }
#pragma unroll
  for (int q = 0; q < 4; ++q) {
    int n = nquad * 64 + q * 16 + lm;
    float bias = b1[n];
#pragma unroll
    for (int r = 0; r < 8; ++r) {
      int m = m0 + r + hh * 8;               // C layout: M = r + 8*(lane/16)
      cur1[(size_t)m * NH + n] = acc[q][r] + bias;
    }
  }
}

// ------------------- phase 2: layer-1 scan (parallel over B*H, serial in t)
__global__ void __launch_bounds__(256)
k_scan1(const float* __restrict__ cur1, const float* __restrict__ alpha1,
        uint16_t* __restrict__ spk1, float* __restrict__ out) {
  __shared__ float red[256];
  int b = blockIdx.x >> 4;                      // 16 blocks per batch row
  int h = ((blockIdx.x & 15) << 8) + threadIdx.x;
  float a = alpha1[h];
  float syn = 0.f, mem = 0.f;
  float* trace1 = out;                          // [T, NH]
  float* memtr  = out + (size_t)T_STEPS * NH;   // [T]
  bool isB0 = (b == 0);

  for (int t = 0; t < T_STEPS; ++t) {
    float cur   = cur1[((size_t)t * BATCH + b) * NH + h];
    float reset = (mem > 1.0f) ? 1.0f : 0.0f;   // H(mem_prev - 1)
    syn = a * syn + cur;
    mem = a * mem + syn - reset;
    float spk = (mem > 1.0f) ? 1.0f : 0.0f;
    spk1[((size_t)t * BATCH + b) * NH + h] = (spk > 0.5f) ? (uint16_t)0x3F80u : (uint16_t)0u;
    if (isB0) {                                  // block-uniform branch
      trace1[(size_t)t * NH + h] = spk;
      red[threadIdx.x] = mem;
      __syncthreads();
      for (int s = 128; s > 0; s >>= 1) {
        if (threadIdx.x < (unsigned)s) red[threadIdx.x] += red[threadIdx.x + s];
        __syncthreads();
      }
      if (threadIdx.x == 0) atomicAdd(&memtr[t], red[0] * (1.0f / NH));
      __syncthreads();
    }
  }
}

// ---- phase 3 (per step t): cur2 = [spk1_t | spk2_prev] @ Wrec^T + brec.
// Block = one 16-row M-tile shared by 8 waves (8 N-tiles). The shared A rows
// are staged into LDS with GLOBAL_LOAD_ASYNC_TO_LDS_B128 in 3 chunks of 48
// K-tiles; waves read A fragments from LDS, B streams from the packed Wrec.
__global__ void __launch_bounds__(256)
k_gemm2(const uint16_t* __restrict__ spk1, const uint16_t* __restrict__ wrp,
        const float* __restrict__ brec, const float* __restrict__ alpha2,
        float* __restrict__ syn2, float* __restrict__ mem2,
        uint16_t* __restrict__ spk2buf, float* __restrict__ rec_out, int t) {
  __shared__ uint16_t sA[CHUNK * 16 * 32];       // 48 KB: [ktl][row16][32 halfs]
  int tid  = threadIdx.x;
  int wave = tid >> 5;
  int lane = tid & 31;
  int blk  = blockIdx.x;          // 0..31
  int mt = blk >> 2;              // 0..7 : batch tile (shared by block)
  int nt = (blk & 3) * 8 + wave;  // 0..31: output tile (per wave)
  int m0 = mt * 16;
  int lm = lane & 15;
  int hh = (lane >> 4) & 1;

  const uint16_t* aSpkBase = spk1 + (size_t)t * BATCH * NH;      // + row*NH
  const uint16_t* rdBuf = spk2buf + (size_t)((t + 1) & 1) * (BATCH * NO);
  uint16_t*       wrBuf = spk2buf + (size_t)(t & 1) * (BATCH * NO);

  uint32_t ldsBase = (uint32_t)(uintptr_t)(lds_u16*)sA;

  v8f acc;
#pragma unroll
  for (int r = 0; r < 8; ++r) acc[r] = 0.0f;

  for (int c = 0; c < 3; ++c) {
    // --- async-stage chunk c of A rows: 48 kt * 16 rows * 64 B = 48 KB
#pragma unroll
    for (int j = 0; j < 12; ++j) {
      int id  = j * 256 + tid;       // 0..3071 transfers of 16 B
      int ktl = id >> 6;             // 64 transfers per K-tile
      int rem = id & 63;
      int row = rem >> 2;
      int seg = rem & 3;
      int ktg = c * CHUNK + ktl;
      const uint16_t* g = (ktg < 2 * KT1)
          ? aSpkBase + (size_t)(m0 + row) * NH + ktg * 32 + seg * 8
          : rdBuf    + (size_t)(m0 + row) * NO + (ktg - 2 * KT1) * 32 + seg * 8;
      uint32_t dst = ldsBase + (uint32_t)(ktl * 1024 + row * 64 + seg * 16);
      async_copy16(dst, g);
    }
    wait_async0();
    __syncthreads();

    // --- consume chunk: A from LDS, B from global (double-buffered)
    Frag16 bC, bN;
    {
      const uint4* pb = (const uint4*)(wrp + (((size_t)nt * KT2 + c * CHUNK) * 32 + lane) * 16);
      bC.u[0] = pb[0]; bC.u[1] = pb[1];
    }
    for (int ktl = 0; ktl < CHUNK; ++ktl) {
      int ktn = (ktl + 1 < CHUNK) ? ktl + 1 : ktl;
      const uint4* pb = (const uint4*)(wrp + (((size_t)nt * KT2 + c * CHUNK + ktn) * 32 + lane) * 16);
      bN.u[0] = pb[0]; bN.u[1] = pb[1];
      Frag16 a;
      const uint4* pa = (const uint4*)(sA + ktl * 512 + lm * 32 + hh * 8);
      a.u[0] = pa[0];   // K 0-7 / 8-15
      a.u[1] = pa[2];   // K 16-23 / 24-31
      acc = __builtin_amdgcn_wmma_f32_16x16x32_bf16(
          false, a.v, false, bC.v, (short)0, acc, false, false);
      bC = bN;
    }
    __syncthreads();   // protect LDS before next chunk overwrites
  }

  int n = nt * 16 + lm;
  float bias = brec[n];
  float a2 = alpha2[n];
#pragma unroll
  for (int r = 0; r < 8; ++r) {
    int bidx = m0 + r + hh * 8;
    size_t idx = (size_t)bidx * NO + n;
    float cur = acc[r] + bias;
    float s = syn2[idx], mm = mem2[idx];
    float reset = (mm > 1.0f) ? 1.0f : 0.0f;
    s = a2 * s + cur;
    mm = a2 * mm + s - reset;
    float spk = (mm > 1.0f) ? 1.0f : 0.0f;
    syn2[idx] = s; mem2[idx] = mm;
    wrBuf[idx] = (spk > 0.5f) ? (uint16_t)0x3F80u : (uint16_t)0u;
    rec_out[((size_t)t * BATCH + bidx) * NO + n] = spk;
  }
}

// --------------------------------------------------------------------------
extern "C" void kernel_launch(void* const* d_in, const int* in_sizes, int n_in,
                              void* d_out, int out_size, void* d_ws, size_t ws_size,
                              hipStream_t stream) {
  const float* x      = (const float*)d_in[0];
  const float* W1     = (const float*)d_in[1];
  const float* b1     = (const float*)d_in[2];
  const float* Wrec   = (const float*)d_in[3];
  const float* brec   = (const float*)d_in[4];
  const float* alpha1 = (const float*)d_in[5];
  const float* alpha2 = (const float*)d_in[6];
  float* out = (float*)d_out;

  char* ws = (char*)d_ws;
  size_t off = 0;
  auto alloc = [&](size_t bytes) -> void* {
    void* p = ws + off;
    off += (bytes + 255) & ~(size_t)255;
    return p;
  };
  uint16_t* xbf  = (uint16_t*)alloc((size_t)T_STEPS * BATCH * NIN * 2); //  64 MB
  uint16_t* w1p  = (uint16_t*)alloc((size_t)NH * NIN * 2);              //  16 MB
  uint16_t* wrp  = (uint16_t*)alloc((size_t)NO * NREC * 2);             // 4.5 MB
  float*    cur1 = (float*)   alloc((size_t)T_STEPS * BATCH * NH * 4);  // 256 MB
  uint16_t* spk1 = (uint16_t*)alloc((size_t)T_STEPS * BATCH * NH * 2);  // 128 MB
  uint16_t* spk2 = (uint16_t*)alloc((size_t)2 * BATCH * NO * 2);        // 256 KB
  float*    syn2 = (float*)   alloc((size_t)BATCH * NO * 4);
  float*    mem2 = (float*)   alloc((size_t)BATCH * NO * 4);

  int nx = T_STEPS * BATCH * NIN;
  k_f32_to_bf16<<<4096, 256, 0, stream>>>(x, xbf, nx);
  k_pack_b<<<(NH * NIN + 255) / 256, 256, 0, stream>>>(W1, w1p, KT1, NH * NIN, NIN);
  k_pack_b<<<(NO * NREC + 255) / 256, 256, 0, stream>>>(Wrec, wrp, KT2, NO * NREC, NREC);
  k_init<<<(BATCH * NO + 255) / 256, 256, 0, stream>>>(
      syn2, mem2, (uint32_t*)spk2, out + (size_t)T_STEPS * NH);

  // phase 1: one big GEMM [16384 x 2048] @ [2048 x 4096]
  k_gemm1<<<8192, 256, 0, stream>>>(xbf, w1p, b1, cur1);
  // phase 2: layer-1 temporal scan, parallel over B*H
  k_scan1<<<(BATCH * NH) / 256, 256, 0, stream>>>(cur1, alpha1, spk1, out);
  // phase 3: sequential recurrent layer, one launch per step
  float* rec_out = out + (size_t)T_STEPS * NH + T_STEPS;
  for (int t = 0; t < T_STEPS; ++t) {
    k_gemm2<<<32, 256, 0, stream>>>(spk1, wrp, brec, alpha2,
                                    syn2, mem2, spk2, rec_out, t);
  }
}
